// MultiHeadLatentAttention_4475355922511
// MI455X (gfx1250) — matrix operand order repaired and checked by reading, not compiled, not run on hardware
//
#include <hip/hip_runtime.h>
#include <hip/hip_bf16.h>
#include <cstdint>
#include <cstddef>

// ---------------- problem constants ----------------
#define B_    2
#define S_    2048
#define H_    2048
#define NH_   16
#define NOPE_ 128
#define ROPE_ 64
#define VD_   128
#define QR_   1024
#define KVR_  512
#define BS_   (B_ * S_)       // 4096 tokens
#define DQK_  (NOPE_ + ROPE_) // 192

typedef __attribute__((ext_vector_type(16))) __bf16 v16bf;
typedef __attribute__((ext_vector_type(8)))  __bf16 v8bf;
typedef __attribute__((ext_vector_type(8)))  float  v8f;

__device__ __forceinline__ unsigned short f32_to_bf16(float f) {
  return (unsigned short)((__builtin_bit_cast(unsigned int, f) + 0x8000u) >> 16);
}
__device__ __forceinline__ unsigned int pack_bf16(float f0, float f1) {
  const unsigned int u0 = __builtin_bit_cast(unsigned int, f0) + 0x8000u;
  const unsigned int u1 = __builtin_bit_cast(unsigned int, f1) + 0x8000u;
  return (u0 >> 16) | (u1 & 0xFFFF0000u);
}

// A fragment, contiguous k order (B loads match): lanes 0-15 hold K={0..15}, lanes 16-31 K={16..31}.
// Both 8-element chunks are contiguous 16B -> two b128 loads + register shuffle.
__device__ __forceinline__ v16bf frag_cat(const unsigned short* p0,
                                          const unsigned short* p1) {
  v8bf lo = *(const v8bf*)p0;
  v8bf hi = *(const v8bf*)p1;
  return __builtin_shufflevector(lo, hi, 0, 1, 2, 3, 4, 5, 6, 7,
                                 8, 9, 10, 11, 12, 13, 14, 15);
}

// ---------------- f32 -> bf16 bulk convert (activations) ----------------
__global__ __launch_bounds__(256) void f32_to_bf16_kernel(
    const float* __restrict__ in, unsigned short* __restrict__ out, int n) {
  int i = (blockIdx.x * 256 + threadIdx.x) * 4;
#pragma unroll
  for (int j = 0; j < 4; ++j)
    if (i + j < n) out[i + j] = f32_to_bf16(in[i + j]);
}

// ---------------- weight convert + transpose: W[K][N] f32 -> Wt[N][K] bf16 --
// One thread per 8-element k-chunk of one output row n. Reads coalesced
// (consecutive threads -> consecutive n), writes one 16B chunk.
__global__ __launch_bounds__(256) void transpose_bf16_kernel(
    const float* __restrict__ W, unsigned short* __restrict__ Wt,
    int K, int N, int total) {
  const int idx = blockIdx.x * 256 + threadIdx.x;  // kc * N + n
  if (idx >= total) return;
  const int n = idx % N;
  const int kc = idx / N;                          // 0 .. K/8-1
  unsigned int w[4];
#pragma unroll
  for (int i = 0; i < 4; ++i) {
    const float f0 = W[(size_t)(kc * 8 + 2 * i) * N + n];
    const float f1 = W[(size_t)(kc * 8 + 2 * i + 1) * N + n];
    w[i] = pack_bf16(f0, f1);
  }
  uint4 u; u.x = w[0]; u.y = w[1]; u.z = w[2]; u.w = w[3];
  *(uint4*)(Wt + (size_t)n * K + kc * 8) = u;
}

// ---------------- rmsnorm (block per row) -> bf16 ----------------
__global__ __launch_bounds__(256) void rmsnorm_bf16(
    const float* __restrict__ X, const float* __restrict__ scale,
    unsigned short* __restrict__ Y, int D) {
  const int row = blockIdx.x;
  const float* xr = X + (size_t)row * D;
  float ss = 0.f;
  for (int i = threadIdx.x; i < D; i += 256) { float v = xr[i]; ss += v * v; }
  __shared__ float red[8];
#pragma unroll
  for (int off = 16; off > 0; off >>= 1) ss += __shfl_xor(ss, off, 32);
  if ((threadIdx.x & 31) == 0) red[threadIdx.x >> 5] = ss;
  __syncthreads();
  if (threadIdx.x == 0) {
    float t = 0.f;
    for (int i = 0; i < 8; ++i) t += red[i];
    red[0] = t;
  }
  __syncthreads();
  const float rs = rsqrtf(red[0] / (float)D + 1e-6f);
  for (int i = threadIdx.x; i < D; i += 256)
    Y[(size_t)row * D + i] = f32_to_bf16(xr[i] * rs * scale[i]);
}

// ---------------- bf16 WMMA GEMM (double-buffered, all-b128 traffic) -------
// C[M,N] (f32) = A[M,K] (bf16 row-major) * Bt[N][K] (bf16, pre-transposed)
// 128x128 tile, BK=32, 256 threads = 8 waves, wave tile 32x64.
// Both LDS tiles are [row][k] so every fragment load is contiguous b128.
__global__ __launch_bounds__(256) void gemm_bf16_wmma(
    const unsigned short* __restrict__ A, const unsigned short* __restrict__ Bt,
    float* __restrict__ C, int M, int N, int K) {
  __shared__ alignas(64) unsigned short As[2][128][32];
  __shared__ alignas(64) unsigned short Bst[2][128][32];

  const int tid  = threadIdx.x;
  const int lane = tid & 31;
  const int wave = tid >> 5;
  const int wm = wave & 3;                 // 4 wave-rows of 32
  const int wn = wave >> 2;                // 2 wave-cols of 64
  const int rowBase = blockIdx.y * 128;
  const int colBase = blockIdx.x * 128;

  v8f acc[2][4];
#pragma unroll
  for (int f = 0; f < 2; ++f)
#pragma unroll
    for (int g = 0; g < 4; ++g)
#pragma unroll
      for (int j = 0; j < 8; ++j) acc[f][g][j] = 0.0f;

  const int lr = tid >> 1;                 // 0..127: A row / Bt row (= out col)
  const int lc = (tid & 1) * 16;           // k-half 0 / 16
  const unsigned short* arow = A + (size_t)(rowBase + lr) * K + lc;
  // clamp instead of predicating (branch-free); guarded C store discards.
  const int bcl = (colBase + lr < N) ? (colBase + lr) : (N - 1);
  const unsigned short* brow = Bt + (size_t)bcl * K + lc;

  uint4 ra0, ra1, rb0, rb1;
  auto gload = [&](int k0) {
    ra0 = ((const uint4*)(arow + k0))[0];
    ra1 = ((const uint4*)(arow + k0))[1];
    rb0 = ((const uint4*)(brow + k0))[0];
    rb1 = ((const uint4*)(brow + k0))[1];
  };
  auto lstore = [&](int buf) {
    *(uint4*)&As[buf][lr][lc]      = ra0;
    *(uint4*)&As[buf][lr][lc + 8]  = ra1;
    *(uint4*)&Bst[buf][lr][lc]     = rb0;
    *(uint4*)&Bst[buf][lr][lc + 8] = rb1;
  };

  gload(0);
  lstore(0);
  __syncthreads();

  const int nt = K >> 5;
  for (int kt = 0; kt < nt; ++kt) {
    const int cur = kt & 1;
    if (kt + 1 < nt) gload((kt + 1) << 5);  // issue next-tile loads early

    v16bf a[2], b[4];
    const int kb8 = (lane >> 4) * 8;
    const int k16 = (lane >> 4) * 16;
#pragma unroll
    for (int f = 0; f < 2; ++f) {
      const unsigned short* p = &As[cur][wm * 32 + f * 16 + (lane & 15)][0];
      a[f] = frag_cat(p + k16, p + k16 + 8);
    }
#pragma unroll
    for (int g = 0; g < 4; ++g)
      b[g] = *(const v16bf*)&Bst[cur][wn * 64 + g * 16 + (lane & 15)][k16];
#pragma unroll
    for (int f = 0; f < 2; ++f)
#pragma unroll
      for (int g = 0; g < 4; ++g)
        acc[f][g] = __builtin_amdgcn_wmma_f32_16x16x32_bf16(
            false, a[f], false, b[g], (short)0, acc[f][g], false, false);

    if (kt + 1 < nt) lstore(cur ^ 1);
    __syncthreads();
  }

  // C layout: VGPR j -> rows (lane>=16 ? 8 : 0)+j, N = lane&15
  const int nl = lane & 15;
  const int mo = (lane >> 4) * 8;
#pragma unroll
  for (int f = 0; f < 2; ++f)
#pragma unroll
    for (int g = 0; g < 4; ++g) {
      const int col = colBase + wn * 64 + g * 16 + nl;
      if (col < N) {
#pragma unroll
        for (int j = 0; j < 8; ++j) {
          const int row = rowBase + wm * 32 + f * 16 + mo + j;
          C[(size_t)row * N + col] = acc[f][g][j];
        }
      }
    }
}

// ---------------- rope + pack Q/K (head-major) and V (d-major) ----------------
__global__ __launch_bounds__(256) void rope_pack(
    const float* __restrict__ qn, const float* __restrict__ qr,
    const float* __restrict__ kn, const float* __restrict__ kr,
    const float* __restrict__ vv,
    unsigned short* __restrict__ Q, unsigned short* __restrict__ Kk,
    unsigned short* __restrict__ Vt) {
  const int bs = blockIdx.x;               // 0..B*S-1
  const int b = bs / S_, s = bs % S_;
  __shared__ float cs[32], sn[32];
  if (threadIdx.x < 32) {
    // inv_freq = 10000^(-p/32) = exp(-p * ln(10000)/32)
    float inv = __expf(-(float)threadIdx.x * 0.28782313f);
    float ang = (float)s * inv;
    cs[threadIdx.x] = __cosf(ang);
    sn[threadIdx.x] = __sinf(ang);
  }
  __syncthreads();
  const float att_scale = 0.07216878364870323f;  // 1/sqrt(192), folded into Q

  for (int idx = threadIdx.x; idx < NH_ * DQK_; idx += 256) {
    const int h = idx / DQK_, d = idx % DQK_;
    const size_t dst = (((size_t)(b * NH_ + h)) * S_ + s) * DQK_ + d;
    float qv, kv;
    if (d < NOPE_) {
      qv = qn[(size_t)bs * (NH_ * NOPE_) + h * NOPE_ + d];
      kv = kn[(size_t)bs * (NH_ * NOPE_) + h * NOPE_ + d];
    } else {
      const int r = d - NOPE_;
      const int p = r >> 1;
      const float q1 = qr[(size_t)bs * (NH_ * ROPE_) + h * ROPE_ + 2 * p];
      const float q2 = qr[(size_t)bs * (NH_ * ROPE_) + h * ROPE_ + 2 * p + 1];
      const float k1 = kr[(size_t)bs * ROPE_ + 2 * p];
      const float k2 = kr[(size_t)bs * ROPE_ + 2 * p + 1];
      const float c = cs[p], si = sn[p];
      if ((r & 1) == 0) { qv = q1 * c - q2 * si; kv = k1 * c - k2 * si; }
      else              { qv = q1 * si + q2 * c; kv = k1 * si + k2 * c; }
      kv *= (1.0f / (float)NH_);
    }
    Q[dst]  = f32_to_bf16(qv * att_scale);
    Kk[dst] = f32_to_bf16(kv);
  }
  // V stored d-major: Vt[bh][d][s] so flash PV B-fragments are contiguous.
  for (int idx = threadIdx.x; idx < NH_ * VD_; idx += 256) {
    const int h = idx / VD_, d = idx % VD_;
    Vt[(((size_t)(b * NH_ + h)) * VD_ + d) * S_ + s] =
        f32_to_bf16(vv[(size_t)bs * (NH_ * VD_) + idx]);
  }
}

// ---------------- causal flash attention (bf16 WMMA, double-buffered) -------
// grid: (B*NH, S/64); block: 128 threads = 4 waves, 16 queries per wave.
__global__ __launch_bounds__(128) void flash_attn(
    const unsigned short* __restrict__ Qg, const unsigned short* __restrict__ Kg,
    const unsigned short* __restrict__ Vtg, unsigned short* __restrict__ Og) {
  const int bh = blockIdx.x;
  const int b = bh / NH_, h = bh % NH_;
  const int qt = blockIdx.y * 64;
  const int tid = threadIdx.x;
  const int lane = tid & 31;
  const int wave = tid >> 5;
  const int q0 = qt + wave * 16;

  __shared__ alignas(64) unsigned short Ks[2][32][DQK_];  // 24 KB, [key][d]
  __shared__ alignas(64) unsigned short Vst[2][VD_][32];  // 16 KB, [d][key]
  __shared__ alignas(64) unsigned short Ps[4][16][32];    //  4 KB, per-wave P

  const unsigned short* Qbase = Qg + ((size_t)bh * S_) * DQK_;
  const unsigned short* Kbase = Kg + ((size_t)bh * S_) * DQK_;
  const unsigned short* Vbase = Vtg + (size_t)bh * VD_ * S_;

  // Q fragments: 6 chunks of K=32 over d=192, via 16B vector loads
  v16bf aq[6];
  {
    const int m = q0 + (lane & 15);
    const int kb8 = (lane >> 4) * 16;
    const unsigned short* qp = Qbase + (size_t)m * DQK_;
#pragma unroll
    for (int c = 0; c < 6; ++c)
      aq[c] = frag_cat(qp + c * 32 + kb8, qp + c * 32 + kb8 + 8);
  }

  v8f o[8];
#pragma unroll
  for (int g = 0; g < 8; ++g)
#pragma unroll
    for (int j = 0; j < 8; ++j) o[g][j] = 0.0f;
  float rmax[8], rsum[8];
#pragma unroll
  for (int j = 0; j < 8; ++j) { rmax[j] = -3.0e38f; rsum[j] = 0.0f; }

  // register-staged K/V tile pipeline (32 keys x 192 d, [d][key] V tile)
  uint4 kreg[6], vreg[4];
  auto kvload = [&](int kt) {
#pragma unroll
    for (int rep = 0; rep < 6; ++rep) {
      const int idx = tid + rep * 128;               // 768 chunks of 8 ushort
      const int r = idx / 24, c8 = (idx % 24) * 8;
      kreg[rep] = *(const uint4*)(Kbase + (size_t)(kt + r) * DQK_ + c8);
    }
#pragma unroll
    for (int rep = 0; rep < 4; ++rep) {
      const int idx = tid + rep * 128;               // 512 chunks of 8 ushort
      const int d = idx >> 2, c8 = (idx & 3) * 8;
      vreg[rep] = *(const uint4*)(Vbase + (size_t)d * S_ + kt + c8);
    }
  };
  auto kvstore = [&](int buf) {
#pragma unroll
    for (int rep = 0; rep < 6; ++rep) {
      const int idx = tid + rep * 128;
      const int r = idx / 24, c8 = (idx % 24) * 8;
      *(uint4*)&Ks[buf][r][c8] = kreg[rep];
    }
#pragma unroll
    for (int rep = 0; rep < 4; ++rep) {
      const int idx = tid + rep * 128;
      const int d = idx >> 2, c8 = (idx & 3) * 8;
      *(uint4*)&Vst[buf][d][c8] = vreg[rep];
    }
  };

  const int ntiles = (qt + 64) / 32;       // keys [0, qt+64) cover causal range
  kvload(0);
  kvstore(0);
  __syncthreads();

  for (int t = 0; t < ntiles; ++t) {
    const int cur = t & 1;
    const int kt = t * 32;
    if (t + 1 < ntiles) kvload((t + 1) * 32);  // overlap with compute

    // scores: 16 queries x 32 keys as two 16x16 C tiles
    v8f sc[2];
#pragma unroll
    for (int nh2 = 0; nh2 < 2; ++nh2) {
      v8f s;
#pragma unroll
      for (int j = 0; j < 8; ++j) s[j] = 0.0f;
      const int n = nh2 * 16 + (lane & 15);
      const int kk = (lane >> 4) * 16;
#pragma unroll
      for (int c = 0; c < 6; ++c) {
        const v16bf bk = *(const v16bf*)&Ks[cur][n][c * 32 + kk];
        s = __builtin_amdgcn_wmma_f32_16x16x32_bf16(
            false, aq[c], false, bk, (short)0, s, false, false);
      }
      // causal mask
      const int kglob = kt + n;
      const int mo = (lane >> 4) * 8;
#pragma unroll
      for (int j = 0; j < 8; ++j)
        if (kglob > q0 + mo + j) s[j] = -1.0e9f;
      sc[nh2] = s;
    }

    // online softmax: row stats across 16-lane groups (C layout rows)
    float corr[8];
#pragma unroll
    for (int j = 0; j < 8; ++j) {
      float mx = fmaxf(sc[0][j], sc[1][j]);
#pragma unroll
      for (int off = 8; off > 0; off >>= 1) mx = fmaxf(mx, __shfl_xor(mx, off, 16));
      const float mnew = fmaxf(rmax[j], mx);
      corr[j] = __expf(rmax[j] - mnew);
      rmax[j] = mnew;
      const float p0 = __expf(sc[0][j] - mnew);
      const float p1 = __expf(sc[1][j] - mnew);
      sc[0][j] = p0; sc[1][j] = p1;
      float rs = p0 + p1;
#pragma unroll
      for (int off = 8; off > 0; off >>= 1) rs += __shfl_xor(rs, off, 16);
      rsum[j] = rsum[j] * corr[j] + rs;
    }
#pragma unroll
    for (int g = 0; g < 8; ++g)
#pragma unroll
      for (int j = 0; j < 8; ++j) o[g][j] *= corr[j];

    // re-stripe P: C layout -> A layout via per-wave LDS scratch
    {
      const int n = lane & 15, mo = (lane >> 4) * 8;
#pragma unroll
      for (int j = 0; j < 8; ++j) {
        Ps[wave][mo + j][n]      = f32_to_bf16(sc[0][j]);
        Ps[wave][mo + j][16 + n] = f32_to_bf16(sc[1][j]);
      }
    }
    v16bf ap;
    {
      const int m = lane & 15, kb8 = (lane >> 4) * 16;
      const unsigned short* pp = &Ps[wave][m][0];
      ap = frag_cat(pp + kb8, pp + kb8 + 8);
    }
    // O += P * V   (8 N-fragments of 16 over d=128), contiguous b128 frags
#pragma unroll
    for (int g = 0; g < 8; ++g) {
      const int n = g * 16 + (lane & 15);
      const int kk = (lane >> 4) * 16;
      const v16bf bv = *(const v16bf*)&Vst[cur][n][kk];
      o[g] = __builtin_amdgcn_wmma_f32_16x16x32_bf16(
          false, ap, false, bv, (short)0, o[g], false, false);
    }

    if (t + 1 < ntiles) kvstore(cur ^ 1);
    __syncthreads();
  }

  // normalize + store attn-out as bf16 in [B,S,NH*VD] (A operand of final GEMM)
  {
    const int n = lane & 15, mo = (lane >> 4) * 8;
#pragma unroll
    for (int g = 0; g < 8; ++g)
#pragma unroll
      for (int j = 0; j < 8; ++j) {
        const int qrow = q0 + mo + j;
        const float val = o[g][j] / rsum[j];
        Og[((size_t)(b * S_ + qrow)) * (NH_ * VD_) + h * VD_ + g * 16 + n] =
            f32_to_bf16(val);
      }
  }
}

// ---------------- host orchestration ----------------
extern "C" void kernel_launch(void* const* d_in, const int* in_sizes, int n_in,
                              void* d_out, int out_size, void* d_ws, size_t ws_size,
                              hipStream_t stream) {
  (void)in_sizes; (void)n_in; (void)out_size; (void)ws_size;
  const float* x         = (const float*)d_in[0];
  const float* W_cq      = (const float*)d_in[1];
  const float* q_scale   = (const float*)d_in[2];
  const float* W_dq_nope = (const float*)d_in[3];
  const float* W_dq_rope = (const float*)d_in[4];
  const float* W_ckv     = (const float*)d_in[5];
  const float* kv_scale  = (const float*)d_in[6];
  const float* W_dk_nope = (const float*)d_in[7];
  const float* W_dv      = (const float*)d_in[8];
  const float* W_krope   = (const float*)d_in[9];
  const float* W_o       = (const float*)d_in[10];
  float* out = (float*)d_out;

  char* base = (char*)d_ws;
  size_t off = 0;
  auto take = [&](size_t bytes) -> void* {
    void* p = base + off;
    off = (off + bytes + 255) & ~(size_t)255;
    return p;
  };
  unsigned short* xb  = (unsigned short*)take((size_t)BS_ * H_ * 2);
  float*  cq          = (float*)take((size_t)BS_ * QR_ * 4);
  unsigned short* nq  = (unsigned short*)take((size_t)BS_ * QR_ * 2);
  float*  qn          = (float*)take((size_t)BS_ * NH_ * NOPE_ * 4);
  float*  qr          = (float*)take((size_t)BS_ * NH_ * ROPE_ * 4);
  float*  ckv         = (float*)take((size_t)BS_ * KVR_ * 4);
  unsigned short* nkv = (unsigned short*)take((size_t)BS_ * KVR_ * 2);
  float*  kn          = (float*)take((size_t)BS_ * NH_ * NOPE_ * 4);
  float*  vv          = (float*)take((size_t)BS_ * NH_ * VD_ * 4);
  float*  kr          = (float*)take((size_t)BS_ * ROPE_ * 4);
  unsigned short* Qb  = (unsigned short*)take((size_t)B_ * NH_ * S_ * DQK_ * 2);
  unsigned short* Kb  = (unsigned short*)take((size_t)B_ * NH_ * S_ * DQK_ * 2);
  unsigned short* Vb  = (unsigned short*)take((size_t)B_ * NH_ * S_ * VD_ * 2);
  unsigned short* ao  = (unsigned short*)take((size_t)BS_ * NH_ * VD_ * 2);
  // pre-transposed bf16 weights: Wt[N][K]
  unsigned short* tcq = (unsigned short*)take((size_t)QR_ * H_ * 2);
  unsigned short* tqn = (unsigned short*)take((size_t)(NH_ * NOPE_) * QR_ * 2);
  unsigned short* tqr = (unsigned short*)take((size_t)(NH_ * ROPE_) * QR_ * 2);
  unsigned short* tkv = (unsigned short*)take((size_t)KVR_ * H_ * 2);
  unsigned short* tkn = (unsigned short*)take((size_t)(NH_ * NOPE_) * KVR_ * 2);
  unsigned short* tdv = (unsigned short*)take((size_t)(NH_ * VD_) * KVR_ * 2);
  unsigned short* tkr = (unsigned short*)take((size_t)ROPE_ * H_ * 2);
  unsigned short* two = (unsigned short*)take((size_t)H_ * (NH_ * VD_) * 2);

  const dim3 t256(256), t128(128);
  auto tposed = [&](const float* W, unsigned short* Wt, int K, int N) {
    const int total = N * (K / 8);
    transpose_bf16_kernel<<<dim3((total + 255) / 256), t256, 0, stream>>>(
        W, Wt, K, N, total);
  };

  f32_to_bf16_kernel<<<dim3((BS_ * H_) / 1024), t256, 0, stream>>>(x, xb, BS_ * H_);
  tposed(W_cq,      tcq, H_,  QR_);
  tposed(W_dq_nope, tqn, QR_, NH_ * NOPE_);
  tposed(W_dq_rope, tqr, QR_, NH_ * ROPE_);
  tposed(W_ckv,     tkv, H_,  KVR_);
  tposed(W_dk_nope, tkn, KVR_, NH_ * NOPE_);
  tposed(W_dv,      tdv, KVR_, NH_ * VD_);
  tposed(W_krope,   tkr, H_,  ROPE_);
  tposed(W_o,       two, NH_ * VD_, H_);

  gemm_bf16_wmma<<<dim3(QR_ / 128, BS_ / 128), t256, 0, stream>>>(
      xb, tcq, cq, BS_, QR_, H_);
  rmsnorm_bf16<<<dim3(BS_), t256, 0, stream>>>(cq, q_scale, nq, QR_);
  gemm_bf16_wmma<<<dim3((NH_ * NOPE_) / 128, BS_ / 128), t256, 0, stream>>>(
      nq, tqn, qn, BS_, NH_ * NOPE_, QR_);
  gemm_bf16_wmma<<<dim3((NH_ * ROPE_) / 128, BS_ / 128), t256, 0, stream>>>(
      nq, tqr, qr, BS_, NH_ * ROPE_, QR_);

  gemm_bf16_wmma<<<dim3(KVR_ / 128, BS_ / 128), t256, 0, stream>>>(
      xb, tkv, ckv, BS_, KVR_, H_);
  rmsnorm_bf16<<<dim3(BS_), t256, 0, stream>>>(ckv, kv_scale, nkv, KVR_);
  gemm_bf16_wmma<<<dim3((NH_ * NOPE_) / 128, BS_ / 128), t256, 0, stream>>>(
      nkv, tkn, kn, BS_, NH_ * NOPE_, KVR_);
  gemm_bf16_wmma<<<dim3((NH_ * VD_) / 128, BS_ / 128), t256, 0, stream>>>(
      nkv, tdv, vv, BS_, NH_ * VD_, KVR_);
  gemm_bf16_wmma<<<dim3(1, BS_ / 128), t256, 0, stream>>>(
      xb, tkr, kr, BS_, ROPE_, H_);

  rope_pack<<<dim3(BS_), t256, 0, stream>>>(qn, qr, kn, kr, vv, Qb, Kb, Vb);

  flash_attn<<<dim3(B_ * NH_, S_ / 64), t128, 0, stream>>>(Qb, Kb, Vb, ao);

  gemm_bf16_wmma<<<dim3(H_ / 128, BS_ / 128), t256, 0, stream>>>(
      ao, two, out, BS_, H_, NH_ * VD_);
}
